// PointConvDensitySetAbstraction_59811714564649
// MI455X (gfx1250) — matrix operand order repaired and checked by reading, not compiled, not run on hardware
//
#include <hip/hip_runtime.h>
#include <hip/hip_bf16.h>
#include <math.h>

typedef __attribute__((ext_vector_type(16))) _Float16 v16h;
typedef __attribute__((ext_vector_type(8)))  _Float16 v8h;
typedef __attribute__((ext_vector_type(8)))  float    v8f;

#define Bn        8
#define Nn        2048
#define DFEAT     64
#define NPOINT    512
#define NSAMPLE   32
#define NPTS      (Bn*NPOINT*NSAMPLE)   /* 131072 */
#define KPAD0     96                    /* 67 padded to mult of 32 */
#define C1        64
#define C2        64
#define C3        128
#define WN        16
#define LINK      2048
#define EPSC      1e-5f
#define BW        0.1f

// ---------------- workspace layout (bytes) ----------------
#define OFF_XYZT    0u          /* f32 [B,N,3]           196608 */
#define OFF_INVD    196608u     /* f32 [B,N]              65536 */
#define OFF_FPS     262144u     /* i32 [B,512]            16384 */
#define OFF_NEWXYZ  278528u     /* f32 [B,512,3]          49152 */
#define OFF_KNN     327680u     /* i32 [B,512,32]        524288 */
#define OFF_DSCALE  851968u     /* f32 [NPTS]            524288 */
#define OFF_WNET    1376256u    /* f16 [4096][16][32]   4194304 */
#define OFF_W1      5570560u    /* f16 [96,64]  swizzled */
#define OFF_W2      5582848u    /* f16 [64,64]  swizzled */
#define OFF_W3      5591040u    /* f16 [64,128] swizzled */
#define OFF_W4      5607424u    /* f16 [2048,128] swizzled */
#define OFF_B1      6131712u
#define OFF_B2      6131968u
#define OFF_B3      6132224u
#define OFF_B4      6132736u
#define OFF_FEAT0   6291456u    /* f16 [NPTS,96]   25165824 */
#define OFF_ACT1    31457280u   /* f16 [NPTS,64]   16777216 */
#define OFF_ACT2    48234496u   /* f16 [NPTS,64]   16777216 */
#define OFF_ACT3    65011712u   /* f16 [4096][128][32] 33554432 */
#define OFF_YAGG    98566144u   /* f16 [4096,2048] 16777216 */

// ---------------- xyz transpose: [B,3,N] -> [B,N,3] ----------------
__global__ void k_transpose_xyz(const float* __restrict__ xyz, float* __restrict__ xyz_t) {
  int i = blockIdx.x * blockDim.x + threadIdx.x;
  if (i >= Bn * Nn) return;
  int b = i / Nn, n = i % Nn;
  const float* src = xyz + (size_t)b * 3 * Nn;
  float* dst = xyz_t + ((size_t)b * Nn + n) * 3;
  dst[0] = src[n]; dst[1] = src[Nn + n]; dst[2] = src[2 * Nn + n];
}

// ---------------- density (all-pairs KDE), write 1/density ----------------
__global__ void k_density(const float* __restrict__ xyz_t, float* __restrict__ inv_dens) {
  __shared__ float sm[256 * 3];
  int b = blockIdx.x >> 3;
  int n = (blockIdx.x & 7) * 256 + threadIdx.x;
  const float* p = xyz_t + ((size_t)b * Nn + n) * 3;
  float px = p[0], py = p[1], pz = p[2];
  const float inv2 = 1.0f / (2.0f * BW * BW);
  float acc = 0.f;
  for (int mc = 0; mc < Nn; mc += 256) {
    __syncthreads();
    const float* q = xyz_t + ((size_t)b * Nn + mc + threadIdx.x) * 3;
    sm[threadIdx.x * 3 + 0] = q[0];
    sm[threadIdx.x * 3 + 1] = q[1];
    sm[threadIdx.x * 3 + 2] = q[2];
    __syncthreads();
    #pragma unroll 8
    for (int j = 0; j < 256; j++) {
      float dx = px - sm[j * 3], dy = py - sm[j * 3 + 1], dz = pz - sm[j * 3 + 2];
      acc += __expf(-(dx * dx + dy * dy + dz * dz) * inv2);
    }
  }
  float dens = acc * (1.0f / (float)Nn) * (1.0f / (2.5f * BW));
  inv_dens[(size_t)b * Nn + n] = 1.0f / dens;
}

// ---------------- farthest point sampling (1 block / batch, LDS resident) ----------------
__global__ void k_fps(const float* __restrict__ xyz_t, int* __restrict__ fps_idx,
                      float* __restrict__ new_xyz, float* __restrict__ out_newxyz_t) {
  __shared__ float sx[Nn], sy[Nn], sz[Nn], dist[Nn];
  __shared__ float rv[256];
  __shared__ int   ri[256];
  __shared__ int   s_far;
  int b = blockIdx.x, t = threadIdx.x;
  for (int i = t; i < Nn; i += 256) {
    const float* p = xyz_t + ((size_t)b * Nn + i) * 3;
    sx[i] = p[0]; sy[i] = p[1]; sz[i] = p[2];
    dist[i] = 1e10f;
  }
  if (t == 0) s_far = 0;
  __syncthreads();
  for (int it = 0; it < NPOINT; it++) {
    int far = s_far;
    float cx = sx[far], cy = sy[far], cz = sz[far];
    if (t == 0) {
      fps_idx[b * NPOINT + it] = far;
      float* nx = new_xyz + ((size_t)b * NPOINT + it) * 3;
      nx[0] = cx; nx[1] = cy; nx[2] = cz;
      out_newxyz_t[(size_t)b * 3 * NPOINT + 0 * NPOINT + it] = cx;
      out_newxyz_t[(size_t)b * 3 * NPOINT + 1 * NPOINT + it] = cy;
      out_newxyz_t[(size_t)b * 3 * NPOINT + 2 * NPOINT + it] = cz;
    }
    float bv = -1.f; int bi = 0;
    for (int i = t; i < Nn; i += 256) {
      float dx = sx[i] - cx, dy = sy[i] - cy, dz = sz[i] - cz;
      float dd = fminf(dist[i], dx * dx + dy * dy + dz * dz);
      dist[i] = dd;
      if (dd > bv) { bv = dd; bi = i; }
    }
    rv[t] = bv; ri[t] = bi;
    __syncthreads();
    for (int s2 = 128; s2 > 0; s2 >>= 1) {
      if (t < s2) {
        if (rv[t + s2] > rv[t] || (rv[t + s2] == rv[t] && ri[t + s2] < ri[t])) {
          rv[t] = rv[t + s2]; ri[t] = ri[t + s2];
        }
      }
      __syncthreads();
    }
    if (t == 0) s_far = ri[0];
    __syncthreads();
  }
}

// ---------------- KNN: 32 smallest distances per centroid ----------------
__global__ void k_knn(const float* __restrict__ xyz_t, const float* __restrict__ new_xyz,
                      int* __restrict__ knn_idx) {
  __shared__ float dist[Nn];
  __shared__ float rv[256];
  __shared__ int   ri[256];
  int g = blockIdx.x;          // b*512+s
  int b = g >> 9;
  int t = threadIdx.x;
  const float* c = new_xyz + (size_t)g * 3;
  float cx = c[0], cy = c[1], cz = c[2];
  for (int i = t; i < Nn; i += 256) {
    const float* p = xyz_t + ((size_t)b * Nn + i) * 3;
    float dx = p[0] - cx, dy = p[1] - cy, dz = p[2] - cz;
    dist[i] = dx * dx + dy * dy + dz * dz;
  }
  __syncthreads();
  for (int kk = 0; kk < NSAMPLE; kk++) {
    float bv = 3.4e38f; int bi = Nn;
    for (int i = t; i < Nn; i += 256) {
      float d = dist[i];
      if (d < bv || (d == bv && i < bi)) { bv = d; bi = i; }
    }
    rv[t] = bv; ri[t] = bi;
    __syncthreads();
    for (int s2 = 128; s2 > 0; s2 >>= 1) {
      if (t < s2) {
        if (rv[t + s2] < rv[t] || (rv[t + s2] == rv[t] && ri[t + s2] < ri[t])) {
          rv[t] = rv[t + s2]; ri[t] = ri[t + s2];
        }
      }
      __syncthreads();
    }
    if (t == 0) { knn_idx[(size_t)g * NSAMPLE + kk] = ri[0]; dist[ri[0]] = 3.4e38f; }
    __syncthreads();
  }
}

// ---------------- gather + pad features, density-net scalar, weight-net ----------------
// wnet stored transposed per group: wnet[g][w][k]  (k contiguous)
__global__ void k_group(const float* __restrict__ xyz_t, const float* __restrict__ points,
                        const float* __restrict__ new_xyz, const int* __restrict__ knn_idx,
                        const float* __restrict__ inv_dens,
                        const float* wnW, const float* wnb, const float* wng, const float* wnbe,
                        const float* dnW1, const float* dnb1, const float* dng1, const float* dnbe1,
                        const float* dnW2, const float* dnb2, const float* dng2, const float* dnbe2,
                        _Float16* __restrict__ feat0, _Float16* __restrict__ wnet,
                        float* __restrict__ dscale) {
  int warp = threadIdx.x >> 5, lane = threadIdx.x & 31;
  int g = blockIdx.x * 8 + warp;    // b*512+s
  int b = g >> 9;
  int row = g * NSAMPLE + lane;     // lane == k
  int idx = knn_idx[row];
  const float* pt = xyz_t + ((size_t)b * Nn + idx) * 3;
  const float* c  = new_xyz + (size_t)g * 3;
  float gx = pt[0] - c[0], gy = pt[1] - c[1], gz = pt[2] - c[2];
  _Float16* f = feat0 + (size_t)row * KPAD0;
  f[0] = (_Float16)gx; f[1] = (_Float16)gy; f[2] = (_Float16)gz;
  const float* pb = points + (size_t)b * DFEAT * Nn + idx;
  #pragma unroll 8
  for (int ch = 0; ch < DFEAT; ch++) f[3 + ch] = (_Float16)pb[(size_t)ch * Nn];
  #pragma unroll
  for (int ch = 3 + DFEAT; ch < KPAD0; ch++) f[ch] = (_Float16)0.f;
  // ---- density scalar: max over k (whole wave = one group), then 1->16->1 net
  float gi = inv_dens[(size_t)b * Nn + idx];
  float mx = gi;
  for (int off = 16; off > 0; off >>= 1) mx = fmaxf(mx, __shfl_xor(mx, off, 32));
  float dn = gi / mx;
  float acc2 = 0.f;
  const float rs = rsqrtf(1.f + EPSC);
  #pragma unroll
  for (int o = 0; o < 16; o++) {
    float h = fmaxf(dng1[o] * rs * (dnW1[o] * dn + dnb1[o]) + dnbe1[o], 0.f);
    acc2 += dnW2[o] * h;
  }
  dscale[row] = fmaxf(dng2[0] * rs * (acc2 + dnb2[0]) + dnbe2[0], 0.f);
  // ---- weight net 3->16 (transposed store: [g][o][k])
  #pragma unroll
  for (int o = 0; o < 16; o++) {
    float a = wnW[o * 3] * gx + wnW[o * 3 + 1] * gy + wnW[o * 3 + 2] * gz + wnb[o];
    wnet[(size_t)g * (WN * NSAMPLE) + o * NSAMPLE + lane] =
        (_Float16)fmaxf(wng[o] * rs * a + wnbe[o], 0.f);
  }
}

// ---------------- fold BN into weights, store in WMMA B-fragment order ----------------
// element e -> [kc][tn][lane][ii] : each lane's 16 halves contiguous (32B)
__global__ void k_prep_w_sw(const float* __restrict__ W, const float* __restrict__ g,
                            _Float16* __restrict__ out, int Kin, int Kpad, int O) {
  int e = blockIdx.x * blockDim.x + threadIdx.x;
  if (e >= Kpad * O) return;
  int ii   = e & 15;
  int lane = (e >> 4) & 31;
  int tk   = e >> 9;              // kc * nt + tn
  int nt   = O >> 4;
  int kc   = tk / nt, tn = tk - kc * nt;
  int col  = tn * 16 + (lane & 15);
  int kk   = kc * 32 + ((lane >> 4) << 3) + (ii & 7) + ((ii >> 3) << 4);
  float s  = g[col] * rsqrtf(1.f + EPSC);
  out[e] = (_Float16)(kk < Kin ? W[(size_t)col * Kin + kk] * s : 0.f);
}
__global__ void k_prep_b(const float* __restrict__ bc, const float* __restrict__ g,
                         const float* __restrict__ be, float* __restrict__ bout, int O) {
  int o = blockIdx.x * blockDim.x + threadIdx.x;
  if (o >= O) return;
  bout[o] = bc[o] * (g[o] * rsqrtf(1.f + EPSC)) + be[o];
}

// ---------------- WMMA fragment loaders (wave32, ISA 7.12.2) ----------------
__device__ __forceinline__ v16h load_a_frag(const _Float16* __restrict__ A, int lda,
                                            int row, int kbase, int lane) {
  const v8h* p = (const v8h*)(A + (size_t)row * lda + kbase + ((lane >> 4) << 3));
  v8h lo = p[0];           // K kbase..+7 (lanes16-31: +8)
  v8h hi = p[2];           // K kbase+16..+23 (lanes16-31: +24)
  return __builtin_shufflevector(lo, hi, 0, 1, 2, 3, 4, 5, 6, 7,
                                 8, 9, 10, 11, 12, 13, 14, 15);
}
__device__ __forceinline__ v16h load_b_frag_sw(const _Float16* __restrict__ Bsw,
                                               int nt, int kc, int tn, int lane) {
  const v16h* p = (const v16h*)(Bsw + (((size_t)kc * nt + tn) * 32 + lane) * 16);
  return *p;               // 32 contiguous bytes per lane
}

// ---------------- strip GEMM: one wave computes a 16 x N strip ----------------
// A fragment loaded once per k-chunk and reused across all NT n-tiles.
// OUTMODE 0: f16 row-major [M,N];  OUTMODE 1: agg layout [M/32][N][32]
template<int NT, int OUTMODE, bool RSCALE>
__global__ __launch_bounds__(256)
void k_gemm_strip(const _Float16* __restrict__ A, const _Float16* __restrict__ Bsw,
                  const float* __restrict__ bias, const float* __restrict__ rowscale,
                  _Float16* __restrict__ C, int M, int K) {
  const int N = NT * 16;
  int wid  = (blockIdx.x * blockDim.x + threadIdx.x) >> 5;   // m-tile index
  int lane = threadIdx.x & 31;
  if (wid >= (M >> 4)) return;
  int arow = wid * 16 + (lane & 15);
  v8f acc[NT];
  #pragma unroll
  for (int j = 0; j < NT; j++) acc[j] = (v8f){0.f,0.f,0.f,0.f,0.f,0.f,0.f,0.f};
  for (int k = 0; k < K; k += 32) {
    v16h a = load_a_frag(A, K, arow, k, lane);
    #pragma unroll
    for (int j = 0; j < NT; j++) {
      v16h bb = load_b_frag_sw(Bsw, NT, k >> 5, j, lane);
      acc[j] = __builtin_amdgcn_wmma_f32_16x16x32_f16(false, a, false, bb, (short)0,
                                                      acc[j], false, false);
    }
  }
  int rbase = wid * 16 + ((lane >> 4) << 3);
  #pragma unroll
  for (int j = 0; j < NT; j++) {
    int col = j * 16 + (lane & 15);
    float bsv = bias[col];
    #pragma unroll
    for (int v = 0; v < 8; v++) {
      int r = rbase + v;
      float val = fmaxf(acc[j][v] + bsv, 0.f);
      if (RSCALE) val *= rowscale[r];
      if (OUTMODE == 0) {
        C[(size_t)r * N + col] = (_Float16)val;
      } else {
        // [g][c][k] with g=r>>5, k=r&31  (K-contiguous for aggregation A loads)
        C[((size_t)(r >> 5) * N + col) * NSAMPLE + (r & 31)] = (_Float16)val;
      }
    }
  }
}

// ---------------- batched aggregation: per (b,s): Y[c,w] = sum_k Xt[c][k]*Wt[w][k], K=32 ----------------
__global__ __launch_bounds__(256)
void k_agg(const _Float16* __restrict__ Xt, const _Float16* __restrict__ Wt,
           _Float16* __restrict__ Y) {
  int wid  = (blockIdx.x * blockDim.x + threadIdx.x) >> 5;  // exactly 32768 waves
  int lane = threadIdx.x & 31;
  int g  = wid >> 3;        // b*512+s
  int mt = wid & 7;         // channel tile (128/16)
  int cch = mt * 16 + (lane & 15);
  int w   = lane & 15;
  int kb  = (lane >> 4) << 4;
  const v8h* pa = (const v8h*)(Xt + ((size_t)g * C3 + cch) * NSAMPLE + kb);
  v8h alo = pa[0], ahi = pa[1];
  v16h a = __builtin_shufflevector(alo, ahi, 0, 1, 2, 3, 4, 5, 6, 7,
                                   8, 9, 10, 11, 12, 13, 14, 15);
  const v16h* pbv = (const v16h*)(Wt + ((size_t)g * WN + w) * NSAMPLE + ((lane >> 4) << 4));
  v16h bb = *pbv;
  v8f acc = {0.f, 0.f, 0.f, 0.f, 0.f, 0.f, 0.f, 0.f};
  acc = __builtin_amdgcn_wmma_f32_16x16x32_f16(false, a, false, bb, (short)0, acc,
                                               false, false);
  int rb = mt * 16 + ((lane >> 4) << 3);
  #pragma unroll
  for (int v = 0; v < 8; v++)
    Y[(size_t)g * (C3 * WN) + (size_t)(rb + v) * WN + w] = (_Float16)acc[v];
}

// ---------------- final linear GEMM (K=2048), relu + transpose to [B,128,S] f32 ----------------
__global__ __launch_bounds__(256)
void k_gemm_final(const _Float16* __restrict__ A, const _Float16* __restrict__ Bsw,
                  const float* __restrict__ bias, float* __restrict__ out,
                  int M, int N, int K) {
  int wid  = (blockIdx.x * blockDim.x + threadIdx.x) >> 5;
  int lane = threadIdx.x & 31;
  int nt = N >> 4;
  int total = (M >> 4) * nt;
  if (wid >= total) return;
  int tm = wid / nt, tn = wid - tm * nt;
  int arow = tm * 16 + (lane & 15);
  int col  = tn * 16 + (lane & 15);
  v8f acc = {0.f, 0.f, 0.f, 0.f, 0.f, 0.f, 0.f, 0.f};
  for (int k = 0; k < K; k += 32) {
    v16h a  = load_a_frag(A, K, arow, k, lane);
    v16h bb = load_b_frag_sw(Bsw, nt, k >> 5, tn, lane);
    acc = __builtin_amdgcn_wmma_f32_16x16x32_f16(false, a, false, bb, (short)0, acc,
                                                 false, false);
  }
  float bsv = bias[col];
  int rbase = tm * 16 + ((lane >> 4) << 3);
  #pragma unroll
  for (int v = 0; v < 8; v++) {
    int r = rbase + v;          // r = b*512 + s
    int bq = r >> 9, s = r & 511;
    out[(size_t)bq * N * NPOINT + (size_t)col * NPOINT + s] = fmaxf(acc[v] + bsv, 0.f);
  }
}

extern "C" void kernel_launch(void* const* d_in, const int* in_sizes, int n_in,
                              void* d_out, int out_size, void* d_ws, size_t ws_size,
                              hipStream_t stream) {
  const float* xyz    = (const float*)d_in[0];
  const float* points = (const float*)d_in[1];
  const float* mlpW0 = (const float*)d_in[2];
  const float* mlpb0 = (const float*)d_in[3];
  const float* mlpg0 = (const float*)d_in[4];
  const float* mlpe0 = (const float*)d_in[5];
  const float* mlpW1 = (const float*)d_in[6];
  const float* mlpb1 = (const float*)d_in[7];
  const float* mlpg1 = (const float*)d_in[8];
  const float* mlpe1 = (const float*)d_in[9];
  const float* mlpW2 = (const float*)d_in[10];
  const float* mlpb2 = (const float*)d_in[11];
  const float* mlpg2 = (const float*)d_in[12];
  const float* mlpe2 = (const float*)d_in[13];
  const float* wnW   = (const float*)d_in[14];
  const float* wnb   = (const float*)d_in[15];
  const float* wng   = (const float*)d_in[16];
  const float* wnbe  = (const float*)d_in[17];
  const float* dnW1  = (const float*)d_in[18];
  const float* dnb1  = (const float*)d_in[19];
  const float* dng1  = (const float*)d_in[20];
  const float* dnbe1 = (const float*)d_in[21];
  const float* dnW2  = (const float*)d_in[22];
  const float* dnb2  = (const float*)d_in[23];
  const float* dng2  = (const float*)d_in[24];
  const float* dnbe2 = (const float*)d_in[25];
  const float* linW  = (const float*)d_in[26];
  const float* linb  = (const float*)d_in[27];
  const float* bnlg  = (const float*)d_in[28];
  const float* bnlbe = (const float*)d_in[29];

  char* ws = (char*)d_ws;
  float*    xyz_t   = (float*)(ws + OFF_XYZT);
  float*    invd    = (float*)(ws + OFF_INVD);
  int*      fpsidx  = (int*)(ws + OFF_FPS);
  float*    newxyz  = (float*)(ws + OFF_NEWXYZ);
  int*      knnidx  = (int*)(ws + OFF_KNN);
  float*    dscale  = (float*)(ws + OFF_DSCALE);
  _Float16* wnet    = (_Float16*)(ws + OFF_WNET);
  _Float16* W1p     = (_Float16*)(ws + OFF_W1);
  _Float16* W2p     = (_Float16*)(ws + OFF_W2);
  _Float16* W3p     = (_Float16*)(ws + OFF_W3);
  _Float16* W4p     = (_Float16*)(ws + OFF_W4);
  float*    B1p     = (float*)(ws + OFF_B1);
  float*    B2p     = (float*)(ws + OFF_B2);
  float*    B3p     = (float*)(ws + OFF_B3);
  float*    B4p     = (float*)(ws + OFF_B4);
  _Float16* feat0   = (_Float16*)(ws + OFF_FEAT0);
  _Float16* act1    = (_Float16*)(ws + OFF_ACT1);
  _Float16* act2    = (_Float16*)(ws + OFF_ACT2);
  _Float16* act3    = (_Float16*)(ws + OFF_ACT3);   // [4096][128][32]
  _Float16* yagg    = (_Float16*)(ws + OFF_YAGG);

  float* out_newxyz = (float*)d_out;                   // [B,3,512]
  float* out_y      = (float*)d_out + Bn * 3 * NPOINT; // [B,128,512]

  // geometry / sampling pipeline
  k_transpose_xyz<<<64, 256, 0, stream>>>(xyz, xyz_t);
  k_density<<<64, 256, 0, stream>>>(xyz_t, invd);
  k_fps<<<Bn, 256, 0, stream>>>(xyz_t, fpsidx, newxyz, out_newxyz);
  k_knn<<<Bn * NPOINT, 256, 0, stream>>>(xyz_t, newxyz, knnidx);

  // fold BN into GEMM weights, swizzled to B-fragment order
  k_prep_w_sw<<<(KPAD0 * C1 + 255) / 256, 256, 0, stream>>>(mlpW0, mlpg0, W1p, 67, KPAD0, C1);
  k_prep_w_sw<<<(C1 * C2 + 255) / 256, 256, 0, stream>>>(mlpW1, mlpg1, W2p, C1, C1, C2);
  k_prep_w_sw<<<(C2 * C3 + 255) / 256, 256, 0, stream>>>(mlpW2, mlpg2, W3p, C2, C2, C3);
  k_prep_w_sw<<<(LINK * C3 + 255) / 256, 256, 0, stream>>>(linW, bnlg, W4p, LINK, LINK, C3);
  k_prep_b<<<1, 256, 0, stream>>>(mlpb0, mlpg0, mlpe0, B1p, C1);
  k_prep_b<<<1, 256, 0, stream>>>(mlpb1, mlpg1, mlpe1, B2p, C2);
  k_prep_b<<<1, 256, 0, stream>>>(mlpb2, mlpg2, mlpe2, B3p, C3);
  k_prep_b<<<1, 256, 0, stream>>>(linb, bnlg, bnlbe, B4p, C3);

  // gather + small nets
  k_group<<<Bn * NPOINT / 8, 256, 0, stream>>>(xyz_t, points, newxyz, knnidx, invd,
                                               wnW, wnb, wng, wnbe,
                                               dnW1, dnb1, dng1, dnbe1,
                                               dnW2, dnb2, dng2, dnbe2,
                                               feat0, wnet, dscale);

  // MLP GEMMs: one wave per 16-row strip, A fragment reused across n-tiles
  {
    int waves = NPTS / 16;                          // 8192
    int blocks = waves / 8;                         // 1024 (8 waves/block)
    k_gemm_strip<4, 0, false><<<blocks, 256, 0, stream>>>(feat0, W1p, B1p, nullptr,
                                                          act1, NPTS, KPAD0);
    k_gemm_strip<4, 0, false><<<blocks, 256, 0, stream>>>(act1, W2p, B2p, nullptr,
                                                          act2, NPTS, C1);
    k_gemm_strip<8, 1, true><<<blocks, 256, 0, stream>>>(act2, W3p, B3p, dscale,
                                                         act3, NPTS, C2);
  }

  // pointconv aggregation: 32768 waves, one WMMA each (K=32 exactly)
  k_agg<<<(Bn * NPOINT * 8) / 8, 256, 0, stream>>>(act3, wnet, yagg);

  // final linear + BN + relu, transposed f32 output
  {
    int tiles4 = ((Bn * NPOINT) / 16) * (C3 / 16); // 2048
    k_gemm_final<<<tiles4 / 8, 256, 0, stream>>>(yagg, W4p, B4p, out_y,
                                                 Bn * NPOINT, C3, LINK);
  }
}
